// MambaBlock_17523466568293
// MI455X (gfx1250) — matrix operand order repaired and checked by reading, not compiled, not run on hardware
//
#include <hip/hip_runtime.h>
#include <hip/hip_bf16.h>
#include <stdint.h>

// ---------------- problem constants ----------------
#define D_MODEL  1024
#define D_INNER  2048
#define D_STATE  16
#define DT_RANK  64
#define D_CONV   4
#define SEQ_L    2048
#define BATCH    2
#define MROWS    (SEQ_L * BATCH)          // 4096 rows in (b,l) space
#define N_XDBL   (DT_RANK + 2 * D_STATE)  // 96
#define CHUNK    64
#define NCHUNK   (SEQ_L / CHUNK)          // 32
#define NCHAN    (BATCH * D_INNER)        // 4096 scan channels

// ---------------- WMMA types (wave32, gfx1250) ----------------
typedef __attribute__((ext_vector_type(16))) __bf16 bf16x16;
typedef __attribute__((ext_vector_type(8)))  float  f32x8;

static __device__ __forceinline__ unsigned short f2bf(float f) {
    union { float f; unsigned int u; } c; c.f = f;
    unsigned int u = c.u;
    u += 0x7fffu + ((u >> 16) & 1u);      // round-to-nearest-even
    return (unsigned short)(u >> 16);
}
static __device__ __forceinline__ float silu_f(float x) { return x / (1.f + __expf(-x)); }
static __device__ __forceinline__ float softplus_f(float x) {
    return (x > 20.f) ? x : log1pf(__expf(x));
}
// low 32 bits of a flat LDS pointer = LDS byte offset (aperture maps addr[31:0] -> LDS)
static __device__ __forceinline__ unsigned lds_off(const void* p) {
    return (unsigned)(uintptr_t)p;
}
static __device__ __forceinline__ void async_b128_to_lds(unsigned lds_addr, const void* gaddr) {
    asm volatile("global_load_async_to_lds_b128 %0, %1, off"
                 :: "v"(lds_addr), "v"((unsigned long long)(uintptr_t)gaddr)
                 : "memory");
}
static __device__ __forceinline__ void wait_async0() {
    asm volatile("s_wait_asynccnt 0" ::: "memory");
}

// ---------------- bf16 WMMA GEMM ----------------
// C(MxN,f32) = A(MxK,bf16 row-major) @ Bt(NxK,bf16 row-major == B transposed)
// Block: 128 threads = 4 waves; block tile 64x64, wave tile 32x32 (2x2 WMMA), K-step 32.
// Tiles staged global->LDS with GLOBAL_LOAD_ASYNC_TO_LDS_B128 (ASYNCcnt).
#define TM 64
#define TN 64
#define TK 32
#define LDSTR 40   // padded LDS row stride (bf16 elems): 80 B, keeps 16B chunk alignment

template <int EPI>   // 0 = none, 1 = softplus(acc + bias[col])
__global__ __launch_bounds__(128)
void gemm_bf16wmma(const unsigned short* __restrict__ A, int lda,
                   const unsigned short* __restrict__ Bt, int ldbt,
                   float* __restrict__ C, int ldc,
                   int M, int N, int K,
                   const float* __restrict__ bias) {
    __shared__ __align__(16) unsigned short lsA[TM * LDSTR];
    __shared__ __align__(16) unsigned short lsB[TN * LDSTR];  // [n][k]

    const int tid  = threadIdx.x;
    const int lane = tid & 31;
    const int wave = tid >> 5;
    const int m0   = blockIdx.y * TM;
    const int n0   = blockIdx.x * TN;
    const int wr   = (wave >> 1) * 32;
    const int wc   = (wave & 1) * 32;
    const int l16  = lane & 15;
    const int hsel = lane >> 4;

    f32x8 acc[2][2] = {};

    for (int k0 = 0; k0 < K; k0 += TK) {
        // A tile: 64 rows x 32 bf16 = 64B/row = 4 x b128 chunks/row, 256 chunks total
        #pragma unroll
        for (int it = 0; it < 2; it++) {
            int c = tid + it * 128;
            int r = c >> 2, k8 = (c & 3) * 8;
            const unsigned short* gp = A + (size_t)(m0 + r) * lda + (k0 + k8);
            async_b128_to_lds(lds_off(&lsA[r * LDSTR + k8]), gp);
        }
        // B tile (transposed source): rows are N, contiguous along K
        #pragma unroll
        for (int it = 0; it < 2; it++) {
            int c = tid + it * 128;
            int r = c >> 2, k8 = (c & 3) * 8;
            int gn = n0 + r; if (gn >= N) gn = N - 1;   // clamp; junk cols masked at store
            const unsigned short* gp = Bt + (size_t)gn * ldbt + (k0 + k8);
            async_b128_to_lds(lds_off(&lsB[r * LDSTR + k8]), gp);
        }
        wait_async0();
        __syncthreads();

        union { bf16x16 v; uint4 q[2]; } fa[2], fb[2];
        // A frag 16x32: lane<16 -> row l16, K {0..7,16..23}; lane>=16 -> K {8..15,24..31}
        #pragma unroll
        for (int i = 0; i < 2; i++) {
            const unsigned short* p = &lsA[(wr + 16 * i + l16) * LDSTR + (hsel ? 8 : 0)];
            fa[i].q[0] = *(const uint4*)(p);
            fa[i].q[1] = *(const uint4*)(p + 16);
        }
        // B frag 32x16: lane-within-half = column; lanes<16 hold K {0..7,16..23}, lanes>=16 K {8..15,24..31}
        #pragma unroll
        for (int j = 0; j < 2; j++) {
            const unsigned short* p = &lsB[(wc + 16 * j + l16) * LDSTR + hsel * 8];
            fb[j].q[0] = *(const uint4*)(p);
            fb[j].q[1] = *(const uint4*)(p + 16);
        }
        #pragma unroll
        for (int i = 0; i < 2; i++)
            #pragma unroll
            for (int j = 0; j < 2; j++)
                acc[i][j] = __builtin_amdgcn_wmma_f32_16x16x32_bf16(
                    false, fa[i].v, false, fb[j].v, (short)0, acc[i][j], false, false);
        __syncthreads();
    }

    // C/D layout: VGPR e -> row (e + 8*hsel), col l16
    #pragma unroll
    for (int i = 0; i < 2; i++)
        #pragma unroll
        for (int j = 0; j < 2; j++)
            #pragma unroll
            for (int e = 0; e < 8; e++) {
                int row = m0 + wr + 16 * i + e + hsel * 8;
                int col = n0 + wc + 16 * j + l16;
                if (row < M && col < N) {
                    float v = acc[i][j][e];
                    if (EPI == 1) v = softplus_f(v + bias[col]);
                    C[(size_t)row * ldc + col] = v;
                }
            }
}

// ---------------- one-time conversion kernels ----------------
// x (l,b,dm) f32 -> xb_bf (m=b*L+l, dm) bf16   (folds the transpose)
__global__ __launch_bounds__(256)
void cvt_xb_kernel(const float* __restrict__ x, unsigned short* __restrict__ xb) {
    int t = blockIdx.x * 256 + threadIdx.x;      // MROWS*D_MODEL
    int c = t & (D_MODEL - 1);
    int m = t >> 10;
    int l = m & (SEQ_L - 1);
    int b = m >> 11;
    xb[t] = f2bf(x[(size_t)(l * BATCH + b) * D_MODEL + c]);
}
// W (KxN f32) -> Wt (NxK bf16)
__global__ __launch_bounds__(256)
void transpose_cvt_kernel(const float* __restrict__ W, unsigned short* __restrict__ Wt,
                          int K, int N) {
    int t = blockIdx.x * 256 + threadIdx.x;      // K*N
    int n = t % N;
    int k = t / N;
    Wt[(size_t)n * K + k] = f2bf(W[(size_t)k * N + n]);
}
// plain f32 -> bf16
__global__ __launch_bounds__(256)
void cvt_bf16_kernel(const float* __restrict__ in, unsigned short* __restrict__ out, int n) {
    int t = blockIdx.x * 256 + threadIdx.x;
    if (t < n) out[t] = f2bf(in[t]);
}

// ---------------- depthwise causal conv1d(4) + SiLU (emits f32 + bf16) ----------------
__global__ __launch_bounds__(256)
void conv_silu_kernel(const float* __restrict__ xar, const float* __restrict__ Wc,
                      const float* __restrict__ bc,
                      float* __restrict__ u, unsigned short* __restrict__ u_bf) {
    int t = blockIdx.x * 256 + threadIdx.x;          // MROWS*D_INNER
    int d = t & (D_INNER - 1);
    int m = t >> 11;
    int l = m & (SEQ_L - 1);
    int b = m >> 11;
    float acc = bc[d];
    #pragma unroll
    for (int j = 0; j < D_CONV; j++) {
        int li = l - (D_CONV - 1) + j;
        if (li >= 0)
            acc += xar[(size_t)(b * SEQ_L + li) * (2 * D_INNER) + d] * Wc[d * D_CONV + j];
    }
    float v = silu_f(acc);
    u[(size_t)m * D_INNER + d] = v;
    u_bf[(size_t)m * D_INNER + d] = f2bf(v);
}

// ---------------- chunked linear-recurrence scan (3 passes, 32x parallel over L) ----------------
__global__ __launch_bounds__(128)
void scan_pass1(const float* __restrict__ delta, const float* __restrict__ u,
                const float* __restrict__ xdbl, const float* __restrict__ A_log,
                float* __restrict__ Pbuf, float* __restrict__ qbuf) {
    int bc = blockIdx.x;                          // 2*32*16 blocks
    int d = (bc & 15) * 128 + threadIdx.x;
    int chunk = (bc >> 4) & (NCHUNK - 1);
    int b = bc >> 9;
    float Aa[D_STATE], P[D_STATE], q[D_STATE];
    #pragma unroll
    for (int n = 0; n < D_STATE; n++) {
        Aa[n] = -__expf(A_log[d * D_STATE + n]);
        P[n] = 1.f; q[n] = 0.f;
    }
    int lbase = chunk * CHUNK;
    for (int li = 0; li < CHUNK; li++) {
        int m = b * SEQ_L + lbase + li;
        float dt = delta[(size_t)m * D_INNER + d];
        float ut = u[(size_t)m * D_INNER + d];
        float du = dt * ut;
        const float* Bv = &xdbl[(size_t)m * N_XDBL + DT_RANK];
        #pragma unroll
        for (int n = 0; n < D_STATE; n++) {
            float da = __expf(dt * Aa[n]);
            q[n] = da * q[n] + du * Bv[n];
            P[n] *= da;
        }
    }
    size_t o = ((size_t)chunk * NCHAN + b * D_INNER + d) * D_STATE;
    #pragma unroll
    for (int n = 0; n < D_STATE; n++) { Pbuf[o + n] = P[n]; qbuf[o + n] = q[n]; }
}

__global__ __launch_bounds__(128)
void scan_pass2(const float* __restrict__ Pbuf, const float* __restrict__ qbuf,
                float* __restrict__ sInit) {
    int ch = blockIdx.x * 128 + threadIdx.x;      // 0..NCHAN-1
    float s[D_STATE];
    #pragma unroll
    for (int n = 0; n < D_STATE; n++) s[n] = 0.f;
    for (int c = 0; c < NCHUNK; c++) {
        size_t o = ((size_t)c * NCHAN + ch) * D_STATE;
        #pragma unroll
        for (int n = 0; n < D_STATE; n++) {
            sInit[o + n] = s[n];
            s[n] = Pbuf[o + n] * s[n] + qbuf[o + n];
        }
    }
}

__global__ __launch_bounds__(128)
void scan_pass3(const float* __restrict__ delta, const float* __restrict__ u,
                const float* __restrict__ xdbl, const float* __restrict__ A_log,
                const float* __restrict__ Dw, const float* __restrict__ xar,
                const float* __restrict__ sInit, unsigned short* __restrict__ y_bf) {
    int bc = blockIdx.x;
    int d = (bc & 15) * 128 + threadIdx.x;
    int chunk = (bc >> 4) & (NCHUNK - 1);
    int b = bc >> 9;
    float Aa[D_STATE], s[D_STATE];
    size_t oi = ((size_t)chunk * NCHAN + b * D_INNER + d) * D_STATE;
    #pragma unroll
    for (int n = 0; n < D_STATE; n++) {
        Aa[n] = -__expf(A_log[d * D_STATE + n]);
        s[n] = sInit[oi + n];
    }
    float Dd = Dw[d];
    int lbase = chunk * CHUNK;
    for (int li = 0; li < CHUNK; li++) {
        int m = b * SEQ_L + lbase + li;
        float dt = delta[(size_t)m * D_INNER + d];
        float ut = u[(size_t)m * D_INNER + d];
        float du = dt * ut;
        const float* Bv = &xdbl[(size_t)m * N_XDBL + DT_RANK];
        const float* Cv = &xdbl[(size_t)m * N_XDBL + DT_RANK + D_STATE];
        float y = 0.f;
        #pragma unroll
        for (int n = 0; n < D_STATE; n++) {
            float da = __expf(dt * Aa[n]);
            s[n] = da * s[n] + du * Bv[n];
            y += s[n] * Cv[n];
        }
        y += Dd * ut;
        float r = xar[(size_t)m * (2 * D_INNER) + D_INNER + d];  // res half of in-proj
        y *= silu_f(r);
        y_bf[(size_t)m * D_INNER + d] = f2bf(y);
    }
}

// ---------------- launch ----------------
extern "C" void kernel_launch(void* const* d_in, const int* in_sizes, int n_in,
                              void* d_out, int out_size, void* d_ws, size_t ws_size,
                              hipStream_t stream) {
    const float* x      = (const float*)d_in[0];
    const float* W_in   = (const float*)d_in[1];
    const float* W_conv = (const float*)d_in[2];
    const float* b_conv = (const float*)d_in[3];
    const float* W_x    = (const float*)d_in[4];
    const float* W_dt   = (const float*)d_in[5];
    const float* b_dt   = (const float*)d_in[6];
    const float* A_log  = (const float*)d_in[7];
    const float* Dw     = (const float*)d_in[8];
    const float* W_out  = (const float*)d_in[9];
    float* out = (float*)d_out;

    // workspace layout (bytes kept 16B aligned; all sizes are multiples of 16B)
    char* w = (char*)d_ws;
    float* xar   = (float*)w;  w += (size_t)MROWS * 2 * D_INNER * 4;   // in-proj out f32
    float* u     = (float*)w;  w += (size_t)MROWS * D_INNER * 4;
    float* xdbl  = (float*)w;  w += (size_t)MROWS * N_XDBL * 4;
    float* delta = (float*)w;  w += (size_t)MROWS * D_INNER * 4;
    float* Pbuf  = (float*)w;  w += (size_t)NCHUNK * NCHAN * D_STATE * 4;
    float* qbuf  = (float*)w;  w += (size_t)NCHUNK * NCHAN * D_STATE * 4;
    float* sInit = (float*)w;  w += (size_t)NCHUNK * NCHAN * D_STATE * 4;
    unsigned short* xb_bf   = (unsigned short*)w;  w += (size_t)MROWS * D_MODEL * 2;
    unsigned short* u_bf    = (unsigned short*)w;  w += (size_t)MROWS * D_INNER * 2;
    unsigned short* xdbl_bf = (unsigned short*)w;  w += (size_t)MROWS * N_XDBL * 2;
    unsigned short* y_bf    = (unsigned short*)w;  w += (size_t)MROWS * D_INNER * 2;
    unsigned short* Win_t   = (unsigned short*)w;  w += (size_t)(2 * D_INNER) * D_MODEL * 2;
    unsigned short* Wx_t    = (unsigned short*)w;  w += (size_t)N_XDBL * D_INNER * 2;
    unsigned short* Wdt_t   = (unsigned short*)w;  w += (size_t)D_INNER * DT_RANK * 2;
    unsigned short* Wout_t  = (unsigned short*)w;  w += (size_t)D_MODEL * D_INNER * 2;

    dim3 blk(128);

    // 0) precision/layout prep: bf16 activations, bf16 transposed weights
    cvt_xb_kernel<<<(MROWS * D_MODEL) / 256, 256, 0, stream>>>(x, xb_bf);
    transpose_cvt_kernel<<<(D_MODEL * 2 * D_INNER) / 256, 256, 0, stream>>>(
        W_in, Win_t, D_MODEL, 2 * D_INNER);
    transpose_cvt_kernel<<<(D_INNER * N_XDBL) / 256, 256, 0, stream>>>(
        W_x, Wx_t, D_INNER, N_XDBL);
    transpose_cvt_kernel<<<(DT_RANK * D_INNER) / 256, 256, 0, stream>>>(
        W_dt, Wdt_t, DT_RANK, D_INNER);
    transpose_cvt_kernel<<<(D_INNER * D_MODEL) / 256, 256, 0, stream>>>(
        W_out, Wout_t, D_INNER, D_MODEL);

    // 1) x_and_res = xb @ W_in
    gemm_bf16wmma<0><<<dim3((2 * D_INNER) / TN, MROWS / TM), blk, 0, stream>>>(
        xb_bf, D_MODEL, Win_t, D_MODEL, xar, 2 * D_INNER, MROWS, 2 * D_INNER, D_MODEL, nullptr);

    // 2) u = silu(causal depthwise conv(xs) + b_conv)   (f32 + bf16)
    conv_silu_kernel<<<(MROWS * D_INNER) / 256, 256, 0, stream>>>(xar, W_conv, b_conv, u, u_bf);

    // 3) x_dbl = u @ W_x   (N = 96, clamped/guarded)
    gemm_bf16wmma<0><<<dim3((N_XDBL + TN - 1) / TN, MROWS / TM), blk, 0, stream>>>(
        u_bf, D_INNER, Wx_t, D_INNER, xdbl, N_XDBL, MROWS, N_XDBL, D_INNER, nullptr);
    cvt_bf16_kernel<<<(MROWS * N_XDBL + 255) / 256, 256, 0, stream>>>(
        xdbl, xdbl_bf, MROWS * N_XDBL);

    // 4) delta = softplus(x_dbl[:, :64] @ W_dt + b_dt)   (A strided, lda = 96)
    gemm_bf16wmma<1><<<dim3(D_INNER / TN, MROWS / TM), blk, 0, stream>>>(
        xdbl_bf, N_XDBL, Wdt_t, DT_RANK, delta, D_INNER, MROWS, D_INNER, DT_RANK, b_dt);

    // 5) chunked selective scan
    scan_pass1<<<BATCH * NCHUNK * (D_INNER / 128), 128, 0, stream>>>(
        delta, u, xdbl, A_log, Pbuf, qbuf);
    scan_pass2<<<NCHAN / 128, 128, 0, stream>>>(Pbuf, qbuf, sInit);
    scan_pass3<<<BATCH * NCHUNK * (D_INNER / 128), 128, 0, stream>>>(
        delta, u, xdbl, A_log, Dw, xar, sInit, y_bf);

    // 6) out = y @ W_out
    gemm_bf16wmma<0><<<dim3(D_MODEL / TN, MROWS / TM), blk, 0, stream>>>(
        y_bf, D_INNER, Wout_t, D_INNER, out, D_MODEL, MROWS, D_MODEL, D_INNER, nullptr);
}
